// BinaryPathEncoder_13134009991561
// MI455X (gfx1250) — matrix operand order repaired and checked by reading, not compiled, not run on hardware
//
#include <hip/hip_runtime.h>
#include <hip/hip_bf16.h>

// ---------------------------------------------------------------------------
// BinaryPathEncoder on MI455X (gfx1250, wave32, WMMA).
//
// maps[4096,512] starts at ones; for depth d = 0..15:
//   m0 = maps @ W0^T ; m1 = maps @ W1^T
//   maps[r] = bit(r,d)==0 ? m0[r] : bit(r,d)==1 ? m1[r] : maps[r]
// where bit(r,d) = ((unique[r]>>d) >= 2) ? ((unique[r]>>d)&1) : 2.
// Then out[g,:] = maps[mapping[g],:]  (32*2048 rows of 512 f32).
//
// maps is kept in bf16 (ping-pong in d_ws); GEMMs run on
// v_wmma_f32_16x16x32_bf16 with f32 accumulation. Each wave owns a 32x32
// output tile: 8 accumulators, 8 WMMAs + 12 b128 loads per k-chunk, no VALU
// conversions in the steady-state loop.
// ---------------------------------------------------------------------------

typedef __attribute__((ext_vector_type(16))) __bf16 v16bf;
typedef __attribute__((ext_vector_type(8)))  __bf16 v8bf;
typedef __attribute__((ext_vector_type(8)))  float  v8f;

#define NROWS 4096
#define DIM   512

// --- weights f32 -> bf16 (layout [t][j][k], identical to primitives) --------
__global__ void cvt_weights_kernel(const float* __restrict__ W,
                                   __bf16* __restrict__ Wb, int n) {
    int i = blockIdx.x * blockDim.x + threadIdx.x;
    if (i < n) Wb[i] = (__bf16)W[i];
}

// --- maps := ones (bf16 1.0 = 0x3F80, packed two per dword) -----------------
__global__ void init_maps_kernel(unsigned int* __restrict__ M, int ndwords) {
    int i = blockIdx.x * blockDim.x + threadIdx.x;
    if (i < ndwords) M[i] = 0x3F803F80u;
}

// --- one scan step: dual-weight WMMA GEMM + branchless per-row select -------
// One wave32 per 32x32 output tile. Tiles: 128 row-tiles x 16 col-tiles
// = 2048 waves; 8 waves per 256-thread block -> 256 blocks.
__global__ __launch_bounds__(256) void step_kernel(
    const __bf16* __restrict__ Min,    // [4096, 512] bf16
    __bf16*       __restrict__ Mout,   // [4096, 512] bf16
    const __bf16* __restrict__ Wb,     // [2, 512, 512] bf16, Wb[t][j][k]
    const int*    __restrict__ unique, // [4096]
    int depth)
{
    const int lane  = threadIdx.x & 31;
    const int gwave = blockIdx.x * 8 + (threadIdx.x >> 5);
    const int row_tile = gwave >> 4;          // 0..127
    const int col_tile = gwave & 15;          // 0..15
    const int row0 = row_tile * 32;
    const int col0 = col_tile * 32;

    const int half = lane >> 4;               // 0: lanes 0-15, 1: lanes 16-31
    const int l15  = lane & 15;

    // acc[rt][ct][t]: rt = row subtile, ct = col subtile, t = weight (W0/W1)
    v8f acc[2][2][2];
#pragma unroll
    for (int rt = 0; rt < 2; ++rt)
#pragma unroll
        for (int ct = 0; ct < 2; ++ct)
#pragma unroll
            for (int t = 0; t < 2; ++t)
                acc[rt][ct][t] = (v8f){};

    // A (16x32 bf16): lane half=0 holds row M=l15, K = kb+0..15;
    // half=1 holds K = kb+16..31 (matches B's contiguous k order).
    const __bf16* arow0 = Min + (size_t)(row0 + l15) * DIM;
    const __bf16* arow1 = arow0 + (size_t)16 * DIM;
    // B (32x16 bf16, contiguous k order, same as A): lane l15 holds column N as
    // W[col + l15][kb + half*16 .. +15] -- contiguous in row-major W.
    const __bf16* b00 = Wb + (size_t)(col0 + l15) * DIM + half * 16; // W0 ct=0
    const __bf16* b01 = b00 + (size_t)16 * DIM;                      // W0 ct=1
    const __bf16* b10 = b00 + (size_t)DIM * DIM;                     // W1 ct=0
    const __bf16* b11 = b10 + (size_t)16 * DIM;                      // W1 ct=1

    for (int kb = 0; kb < DIM; kb += 32) {
        v16bf a0, a1;
        *(v8bf*)&a0        = *(const v8bf*)(arow0 + kb + half * 16);
        *((v8bf*)&a0 + 1)  = *(const v8bf*)(arow0 + kb + half * 16 + 8);
        *(v8bf*)&a1        = *(const v8bf*)(arow1 + kb + half * 16);
        *((v8bf*)&a1 + 1)  = *(const v8bf*)(arow1 + kb + half * 16 + 8);

        v16bf vb00 = *(const v16bf*)(b00 + kb);
        v16bf vb01 = *(const v16bf*)(b01 + kb);
        v16bf vb10 = *(const v16bf*)(b10 + kb);
        v16bf vb11 = *(const v16bf*)(b11 + kb);

        acc[0][0][0] = __builtin_amdgcn_wmma_f32_16x16x32_bf16(
            false, a0, false, vb00, (short)0, acc[0][0][0], false, false);
        acc[0][0][1] = __builtin_amdgcn_wmma_f32_16x16x32_bf16(
            false, a0, false, vb10, (short)0, acc[0][0][1], false, false);
        acc[0][1][0] = __builtin_amdgcn_wmma_f32_16x16x32_bf16(
            false, a0, false, vb01, (short)0, acc[0][1][0], false, false);
        acc[0][1][1] = __builtin_amdgcn_wmma_f32_16x16x32_bf16(
            false, a0, false, vb11, (short)0, acc[0][1][1], false, false);
        acc[1][0][0] = __builtin_amdgcn_wmma_f32_16x16x32_bf16(
            false, a1, false, vb00, (short)0, acc[1][0][0], false, false);
        acc[1][0][1] = __builtin_amdgcn_wmma_f32_16x16x32_bf16(
            false, a1, false, vb10, (short)0, acc[1][0][1], false, false);
        acc[1][1][0] = __builtin_amdgcn_wmma_f32_16x16x32_bf16(
            false, a1, false, vb01, (short)0, acc[1][1][0], false, false);
        acc[1][1][1] = __builtin_amdgcn_wmma_f32_16x16x32_bf16(
            false, a1, false, vb11, (short)0, acc[1][1][1], false, false);
    }

    // C/D layout: VGPR r, lanes 0-15 -> (M=r, N=lane); lanes 16-31 -> (M=r+8).
    // Branchless select: always fetch the passthrough value (L2-resident).
#pragma unroll
    for (int rt = 0; rt < 2; ++rt) {
#pragma unroll
        for (int r = 0; r < 8; ++r) {
            const int row = row0 + rt * 16 + r + half * 8;
            const int sh  = unique[row] >> depth;
            const bool active = (sh >= 2);
            const bool take1  = (sh & 1) != 0;
#pragma unroll
            for (int ct = 0; ct < 2; ++ct) {
                const int col = col0 + ct * 16 + l15;
                const __bf16 pass = Min[(size_t)row * DIM + col];
                const float gem = take1 ? acc[rt][ct][1][r]
                                        : acc[rt][ct][0][r];
                const float val = active ? gem : (float)pass;
                Mout[(size_t)row * DIM + col] = (__bf16)val;
            }
        }
    }
}

// --- final gather: out[g,:] = (float)maps[mapping[g],:] ---------------------
// One block (64 threads) per output row; 8 elements (16B bf16 in, 32B f32 out)
// per lane.
__global__ __launch_bounds__(64) void gather_kernel(
    const __bf16* __restrict__ maps,
    const int*    __restrict__ mapping,
    float*        __restrict__ out)
{
    const int g   = blockIdx.x;
    const int t   = threadIdx.x;
    const int src = mapping[g];

    v8bf v = *(const v8bf*)(maps + (size_t)src * DIM + t * 8);
    float4 o0, o1;
    o0.x = (float)v[0]; o0.y = (float)v[1]; o0.z = (float)v[2]; o0.w = (float)v[3];
    o1.x = (float)v[4]; o1.y = (float)v[5]; o1.z = (float)v[6]; o1.w = (float)v[7];

    float4* d = (float4*)(out + (size_t)g * DIM + t * 8);
    d[0] = o0;
    d[1] = o1;
}

extern "C" void kernel_launch(void* const* d_in, const int* in_sizes, int n_in,
                              void* d_out, int out_size, void* d_ws, size_t ws_size,
                              hipStream_t stream) {
    const int*   unique  = (const int*)d_in[0];    // [4096]
    const int*   mapping = (const int*)d_in[1];    // [32*2048]
    const float* prim    = (const float*)d_in[2];  // [2,512,512]
    float*       out     = (float*)d_out;

    char* ws = (char*)d_ws;
    __bf16* mapsA = (__bf16*)ws;                              // 4 MB
    __bf16* mapsB = (__bf16*)(ws + (size_t)4 * 1024 * 1024);  // 4 MB
    __bf16* Wb    = (__bf16*)(ws + (size_t)8 * 1024 * 1024);  // 1 MB

    const int nW = 2 * DIM * DIM;          // 524288
    const int nMd = NROWS * DIM / 2;       // 1048576 dwords of bf16 ones
    const int nG = in_sizes[1];            // 65536 output rows

    cvt_weights_kernel<<<(nW + 255) / 256, 256, 0, stream>>>(prim, Wb, nW);
    init_maps_kernel<<<(nMd + 255) / 256, 256, 0, stream>>>(
        (unsigned int*)mapsA, nMd);

    __bf16* src = mapsA;
    __bf16* dst = mapsB;
    for (int d = 0; d < 16; ++d) {
        // 2048 tiles / 8 waves per block = 256 blocks
        step_kernel<<<256, 256, 0, stream>>>(src, dst, Wb, unique, d);
        __bf16* t = src; src = dst; dst = t;
    }

    gather_kernel<<<nG, 64, 0, stream>>>(src, mapping, out);
}
